// GCNLayer_63118839382673
// MI455X (gfx1250) — matrix operand order repaired and checked by reading, not compiled, not run on hardware
//
#include <hip/hip_runtime.h>
#include <hip/hip_bf16.h>
#include <stdint.h>

#define N_NODES  50000
#define N_EDGES  800000
#define FEATS    512
#define KEEP_THRESH 3865470566u    /* floor(0.9 * 2^32) */
#define INV_KEEP    (1.0f / 0.9f)
#define LDS_STRIDE  520            /* 512 + 8 bf16 pad: row stride 1040B = 260 dwords
                                      => 260 % 64 = 4 banks shift per row, conflict-free */

typedef __attribute__((ext_vector_type(16))) __bf16 v16bf;
typedef __attribute__((ext_vector_type(8)))  float  v8f;

union FragBF { v16bf v; uint4 u[2]; };

__device__ __forceinline__ unsigned short f32_to_bf16(float f) {
    union { float f; uint32_t u; } c; c.f = f;
    uint32_t u = c.u;
    if ((u & 0x7F800000u) == 0x7F800000u) return (unsigned short)(u >> 16); // inf/nan
    uint32_t r = u + 0x7FFFu + ((u >> 16) & 1u);  // round-to-nearest-even
    return (unsigned short)(r >> 16);
}

__device__ __forceinline__ uint32_t hash_u32(uint32_t x) {
    x ^= x >> 16; x *= 0x7FEB352Du;
    x ^= x >> 15; x *= 0x846CA68Bu;
    x ^= x >> 16;
    return x;
}

// ---------------------------------------------------------------------------
// Kernel 1: W (f32, [K=512][N=512] row-major) -> Wt (bf16, [N=512][K=512])
// ---------------------------------------------------------------------------
__global__ __launch_bounds__(256) void wt_bf16_kernel(
    const float* __restrict__ W, unsigned short* __restrict__ Wt)
{
    const int idx = blockIdx.x * 256 + threadIdx.x;   // 0 .. 512*512-1
    const int n = idx >> 9;
    const int k = idx & 511;
    Wt[(size_t)n * FEATS + k] = f32_to_bf16(W[(size_t)k * FEATS + n]);
}

// ---------------------------------------------------------------------------
// Kernel 2: fused dropout + SpMM segment-sum.
// One 256-thread block per destination node; edge range found by binary
// search in the sorted edge_dst array (no atomics). Each thread owns 2
// contiguous feature columns. Result written as bf16 (WMMA A operand).
// ---------------------------------------------------------------------------
__global__ __launch_bounds__(256) void spmm_dropout_kernel(
    const float* __restrict__ x, const float* __restrict__ edge_w,
    const int* __restrict__ edge_src, const int* __restrict__ edge_dst,
    unsigned short* __restrict__ agg)
{
    const int node = blockIdx.x;

    // lower_bound(node), lower_bound(node+1) — uniform scalar work
    int lo, hi;
    { int l = 0, r = N_EDGES;
      while (l < r) { int m = (l + r) >> 1; if (edge_dst[m] < node) l = m + 1; else r = m; }
      lo = l; }
    { int l = lo, r = N_EDGES;
      while (l < r) { int m = (l + r) >> 1; if (edge_dst[m] <= node) l = m + 1; else r = m; }
      hi = l; }

    const int f = threadIdx.x * 2;
    float acc0 = 0.0f, acc1 = 0.0f;

    for (int e = lo; e < hi; ++e) {
        const int   src = edge_src[e];
        const float w   = edge_w[e];
        if (e + 1 < hi) {
            // warm up the next gathered row (global_prefetch_b8)
            __builtin_prefetch(x + (size_t)edge_src[e + 1] * FEATS + f, 0, 1);
        }
        const float2 xv = *(const float2*)(x + (size_t)src * FEATS + f);
        const uint32_t base = (uint32_t)src * FEATS + (uint32_t)f;
        if (hash_u32(base)      < KEEP_THRESH) acc0 = fmaf(xv.x, w, acc0);
        if (hash_u32(base + 1u) < KEEP_THRESH) acc1 = fmaf(xv.y, w, acc1);
    }

    unsigned short* row = agg + (size_t)node * FEATS;
    row[f]     = f32_to_bf16(acc0 * INV_KEEP);
    row[f + 1] = f32_to_bf16(acc1 * INV_KEEP);
}

// ---------------------------------------------------------------------------
// Kernel 3: out = relu(agg @ W + b) via v_wmma_f32_16x16x32_bf16.
//
// Block = 8 waves, 128x64 C macro-tile. The 64x512 bf16 B panel (64 KB,
// fits easily in the 320 KB WGP LDS) is staged once per block by all 256
// threads, then every wave's B fragments come from ds_load_b128 at LDS
// latency — shared 8 ways, no per-k-step barriers since the panel is
// invariant for the block. Row stride padded to 520 bf16 so the 16 lanes
// reading 16 different n-rows hit disjoint bank groups.
// A fragments stream from global (8x L2 reuse across the grid.y panels),
// double-buffered across k-steps. C stored non-temporal (written once).
// ---------------------------------------------------------------------------
__global__ __launch_bounds__(256) void gemm_wmma_kernel(
    const unsigned short* __restrict__ A,    // agg bf16 [M=50000][K=512]
    const unsigned short* __restrict__ Bt,   // Wt  bf16 [N=512][K=512]
    const float* __restrict__ bias,
    float* __restrict__ out)
{
    __shared__ __align__(16) unsigned short lb[64 * LDS_STRIDE];

    const int tid = threadIdx.x;
    const int n0  = blockIdx.y * 64;

    // Cooperative B-panel load: 64 rows x 512 bf16 = 8192 uint4, 32 per thread.
    #pragma unroll
    for (int it = 0; it < 32; ++it) {
        const int idx = tid + it * 256;          // 0 .. 8191
        const int row = idx >> 7;                // / 128 uint4s per row
        const int c8  = (idx & 127) * 8;         // element column
        const uint4 v = *(const uint4*)(Bt + (size_t)(n0 + row) * FEATS + c8);
        *(uint4*)(lb + row * LDS_STRIDE + c8) = v;
    }
    __syncthreads();

    const int wave  = tid >> 5;
    const int lane  = tid & 31;
    const int m0    = (blockIdx.x * 8 + wave) * 16;
    if (m0 >= N_NODES) return;               // wave-uniform, after the barrier
    const int lmod  = lane & 15;
    const int lhalf = lane >> 4;             // 0: low K-half lanes, 1: high

    v8f acc[4];
    #pragma unroll
    for (int t = 0; t < 4; ++t)
        #pragma unroll
        for (int i = 0; i < 8; ++i) acc[t][i] = 0.0f;

    const unsigned short* arow = A + (size_t)(m0 + lmod) * FEATS;
    const unsigned short* lrow = lb + lmod * LDS_STRIDE;

    // A 16x32 fragment: lanes 0-15 rows m0..m0+15, K={kc..kc+7, kc+16..kc+23};
    // lanes 16-31 same rows, K offset +8. Two b128 loads, double-buffered.
    FragBF aCur, aNxt;
    aCur.u[0] = *(const uint4*)(arow + 0  + lhalf * 8);
    aCur.u[1] = *(const uint4*)(arow + 16 + lhalf * 8);

    #pragma unroll
    for (int kc = 0; kc < FEATS; kc += 32) {
        if (kc + 32 < FEATS) {
            aNxt.u[0] = *(const uint4*)(arow + kc + 32 +      lhalf * 8);
            aNxt.u[1] = *(const uint4*)(arow + kc + 32 + 16 + lhalf * 8);
        }
        #pragma unroll
        for (int t = 0; t < 4; ++t) {
            // B 32x16 fragment from LDS: lane n = lmod, K = {kc+lhalf*8..+7, kc+16+lhalf*8..+7}
            const unsigned short* bp = lrow + t * 16 * LDS_STRIDE + kc + lhalf * 8;
            FragBF bf;
            bf.u[0] = *(const uint4*)(bp);
            bf.u[1] = *(const uint4*)(bp + 16);
            acc[t] = __builtin_amdgcn_wmma_f32_16x16x32_bf16(
                false, aCur.v, false, bf.v, (short)0, acc[t], false, false);
        }
        aCur = aNxt;   // fully unrolled: register renaming, no moves
    }

    // C/D layout: VGPR r -> M = r + 8*lhalf, N = lmod.
    // Non-temporal: 102 MB of C written once; keep A/B panels hot in L2.
    #pragma unroll
    for (int t = 0; t < 4; ++t) {
        const int col = n0 + t * 16 + lmod;
        const float bv = bias[col];
        #pragma unroll
        for (int r = 0; r < 8; ++r) {
            const int row = m0 + lhalf * 8 + r;
            float v = acc[t][r] + bv;
            v = v > 0.0f ? v : 0.0f;
            __builtin_nontemporal_store(v, out + (size_t)row * FEATS + col);
        }
    }
}

// ---------------------------------------------------------------------------
extern "C" void kernel_launch(void* const* d_in, const int* in_sizes, int n_in,
                              void* d_out, int out_size, void* d_ws, size_t ws_size,
                              hipStream_t stream) {
    const float* x        = (const float*)d_in[0];   // [50000, 512]
    const float* edge_w   = (const float*)d_in[1];   // [800000]
    const float* W        = (const float*)d_in[2];   // [512, 512]
    const float* b        = (const float*)d_in[3];   // [512]
    const int*   edge_src = (const int*)d_in[4];     // [800000]
    const int*   edge_dst = (const int*)d_in[5];     // [800000] (sorted)

    unsigned short* agg = (unsigned short*)d_ws;                    // 50000*512 bf16 (51.2 MB)
    unsigned short* Wt  = agg + (size_t)N_NODES * FEATS;            // 512*512 bf16 (0.5 MB), 16B-aligned
    float* out = (float*)d_out;

    wt_bf16_kernel<<<dim3((FEATS * FEATS) / 256), dim3(256), 0, stream>>>(W, Wt);

    spmm_dropout_kernel<<<dim3(N_NODES), dim3(256), 0, stream>>>(
        x, edge_w, edge_src, edge_dst, agg);

    dim3 grid((N_NODES / 16 + 7) / 8, FEATS / 64);  // 391 x 8
    gemm_wmma_kernel<<<grid, dim3(256), 0, stream>>>(agg, Wt, b, out);
}
